// IterAttnConvNeXtIsotropic_83236466197104
// MI455X (gfx1250) — matrix operand order repaired and checked by reading, not compiled, not run on hardware
//
#include <hip/hip_runtime.h>
#include <math.h>

#define DEPTH   18
#define DIM     384
#define HIDDEN  1536
#define NCLS    1000
#define BATCH   32
#define GRID14  14
#define TOK     196            // 14*14
#define NTOK    (BATCH * TOK)  // 6272
#define PATCH_K 768            // 3*16*16
#define LATPAD  64             // latent rows padded to GEMM tiling

typedef unsigned short u16;
typedef __attribute__((ext_vector_type(16))) __bf16 v16bf;
typedef __attribute__((ext_vector_type(8)))  float  v8f;

union FragAB { v16bf v; uint4 q[2]; };

__device__ __forceinline__ u16 f2bf(float f) {
  union { float f; unsigned u; } c; c.f = f;
  unsigned u = c.u;
  u += 0x7FFFu + ((u >> 16) & 1u);     // round-to-nearest-even
  return (u16)(u >> 16);
}

__device__ __forceinline__ float gelu_exact(float x) {
  return 0.5f * x * (1.0f + erff(x * 0.70710678118654752f));
}

// ---------------------------------------------------------------------------
// bf16 WMMA GEMM: C[MxN] = A[MxK] * Bt[NxK]^T  (+bias)(gelu)(+res), fp32 acc
//   A row-major bf16; Bt is B pre-transposed to N-major bf16.
//   Block = 256 threads (8 waves). Block tile = 128M x 64N; wave tile 16M x 64N.
//   B tile (64 x 32 bf16 = 4KB) staged in LDS, double-buffered, shared by all
//   8 waves (8x L2-traffic reduction). A read directly (2 x b128 per k-step).
//   LDS rows padded to 80B so ds_load_b128 fragment reads spread banks.
//   flags: 1=bias, 2=gelu, 4=residual add
// ---------------------------------------------------------------------------
#define BROW_U16 40   // 80 bytes per 32-k row (64B data + 16B pad)

__global__ __launch_bounds__(256) void gemm_bf16_wmma(
    const u16* __restrict__ A, const u16* __restrict__ Bt,
    float* __restrict__ Cf, u16* __restrict__ Cbf,
    const float* __restrict__ bias, const float* __restrict__ res,
    int M, int N, int K, int flags)
{
  __shared__ u16 lbs[2][64 * BROW_U16];   // 2 x 5120 B

  const int tid  = threadIdx.x;
  const int wave = tid >> 5;
  const int lane = tid & 31;
  const int half = lane >> 4;   // 0: lanes 0-15, 1: lanes 16-31
  const int l16  = lane & 15;
  const int m0 = blockIdx.x * 128 + wave * 16;
  const int n0 = blockIdx.y * 64;
  const bool active = (m0 < M);
  const int mbase = active ? m0 : (M - 16);   // clamp: keep loads in-bounds, EXEC all-ones

  v8f acc[4] = {};

  // A frag (16x32 bf16): lane M = l16; elems 0..7 = K[half*8..], 8..15 = K[16+half*8..]
  const u16* arow = A + (size_t)(mbase + l16) * K + half * 8;

  // cooperative B staging: 64 rows x 4 chunks of 16B; one chunk per thread
  const int srow = tid >> 2;
  const int schk = tid & 3;
  const u16* gB = Bt + (size_t)(n0 + srow) * K + schk * 8;
  u16* lB = &lbs[0][0] + srow * BROW_U16 + schk * 8;

  uint4 stage = *(const uint4*)(gB);
  *(uint4*)(lB) = stage;
  __syncthreads();

  for (int k0 = 0; k0 < K; k0 += 32) {
    const int buf = (k0 >> 5) & 1;
    const bool more = (k0 + 32) < K;
    if (more) stage = *(const uint4*)(gB + k0 + 32);   // prefetch next B tile

    FragAB a;
    a.q[0] = *(const uint4*)(arow + k0);        // K run 0..7  (per lane half)
    a.q[1] = *(const uint4*)(arow + k0 + 16);   // K run 16..23 (per lane half)

#pragma unroll
    for (int j = 0; j < 4; ++j) {
      FragAB bf;
      const u16* lp = &lbs[buf][0] + (j * 16 + l16) * BROW_U16 + half * 8;
      bf.q[0] = *(const uint4*)(lp);
      bf.q[1] = *(const uint4*)(lp + 16);
      acc[j] = __builtin_amdgcn_wmma_f32_16x16x32_bf16(
          false, a.v, false, bf.v, (short)0, acc[j], false, false);
    }

    if (more) *(uint4*)(&lbs[buf ^ 1][0] + srow * BROW_U16 + schk * 8) = stage;
    __syncthreads();
  }

  if (!active) return;

#pragma unroll
  for (int j = 0; j < 4; ++j) {
    const int n = n0 + j * 16 + l16;
    const float bv = (flags & 1) ? bias[n] : 0.0f;
#pragma unroll
    for (int r = 0; r < 8; ++r) {
      const int m = m0 + half * 8 + r;          // C layout: VGPR r -> M = r + 8*half
      float v = acc[j][r] + bv;
      if (flags & 2) v = gelu_exact(v);
      const size_t off = (size_t)m * N + n;
      if (flags & 4) v += res[off];
      if (Cf)  Cf[off]  = v;
      if (Cbf) Cbf[off] = f2bf(v);
    }
  }
}

// ---------------------------------------------------------------------------
// Elementwise / data-movement helpers
// ---------------------------------------------------------------------------
__global__ void convert_f32_bf16(const float* __restrict__ in, u16* __restrict__ out, int n) {
  int idx = blockIdx.x * 256 + threadIdx.x;
  if (idx < n) out[idx] = f2bf(in[idx]);
}

// in: (K x N) row-major f32  ->  out: (N x K) row-major bf16
__global__ void transpose_f32_bf16(const float* __restrict__ in, u16* __restrict__ out,
                                   int K, int N) {
  int idx = blockIdx.x * 256 + threadIdx.x;
  if (idx >= K * N) return;
  int k = idx % K;
  int n = idx / K;
  out[(size_t)n * K + k] = f2bf(in[(size_t)k * N + n]);
}

// x (32,3,224,224) -> patches (6272 x 768) bf16, feature = c*256 + iy*16 + ix
__global__ void patchify(const float* __restrict__ x, u16* __restrict__ out) {
  int idx = blockIdx.x * 256 + threadIdx.x;
  if (idx >= NTOK * PATCH_K) return;
  int f = idx % PATCH_K;
  int tok = idx / PATCH_K;
  int b = tok / TOK, p = tok % TOK;
  int py = p / GRID14, px = p % GRID14;
  int c = f >> 8, r = f & 255, iy = r >> 4, ix = r & 15;
  float v = x[(((size_t)b * 3 + c) * 224 + (py * 16 + iy)) * 224 + (px * 16 + ix)];
  out[idx] = f2bf(v);
}

__global__ void init_latent(const float* __restrict__ latent,
                            float* __restrict__ latf, u16* __restrict__ latbf) {
  int idx = blockIdx.x * 256 + threadIdx.x;
  if (idx >= LATPAD * DIM) return;
  int row = idx / DIM, d = idx % DIM;
  float v = (row < BATCH) ? latent[d] : 0.0f;
  latf[idx]  = v;
  latbf[idx] = f2bf(v);
}

// Depthwise 7x7, pad 3, NHWC (tokens x DIM), bias fused
__global__ __launch_bounds__(256) void dwconv7(
    const float* __restrict__ h, const float* __restrict__ w,
    const float* __restrict__ bias, float* __restrict__ y)
{
  int idx = blockIdx.x * 256 + threadIdx.x;
  if (idx >= NTOK * DIM) return;
  int c = idx % DIM;
  int t = idx / DIM;
  int px = t % GRID14; int ty = t / GRID14;
  int py = ty % GRID14; int b = ty / GRID14;
  float s = bias[c];
  const float* wc = w + c * 49;
#pragma unroll
  for (int ky = 0; ky < 7; ++ky) {
    int yy = py + ky - 3;
    if (yy < 0 || yy >= GRID14) continue;
#pragma unroll
    for (int kx = 0; kx < 7; ++kx) {
      int xx = px + kx - 3;
      if (xx < 0 || xx >= GRID14) continue;
      s += h[((size_t)b * TOK + yy * GRID14 + xx) * DIM + c] * wc[ky * 7 + kx];
    }
  }
  y[idx] = s;
}

// LayerNorm over last dim, one row per block (128 threads)
__global__ __launch_bounds__(128) void ln_rows(
    const float* __restrict__ in, const float* __restrict__ g, const float* __restrict__ b,
    float eps, float* __restrict__ outf, u16* __restrict__ outbf, int dim)
{
  const int row = blockIdx.x;
  const int tid = threadIdx.x;
  const float* x = in + (size_t)row * dim;
  __shared__ float red[128];
  __shared__ float s_mu, s_inv;

  float s = 0.f;
  for (int i = tid; i < dim; i += 128) s += x[i];
  red[tid] = s; __syncthreads();
  for (int o = 64; o > 0; o >>= 1) { if (tid < o) red[tid] += red[tid + o]; __syncthreads(); }
  if (tid == 0) s_mu = red[0] / dim;
  __syncthreads();
  const float mu = s_mu;

  float s2 = 0.f;
  for (int i = tid; i < dim; i += 128) { float d = x[i] - mu; s2 += d * d; }
  red[tid] = s2; __syncthreads();
  for (int o = 64; o > 0; o >>= 1) { if (tid < o) red[tid] += red[tid + o]; __syncthreads(); }
  if (tid == 0) s_inv = rsqrtf(red[0] / dim + eps);
  __syncthreads();
  const float inv = s_inv;

  for (int i = tid; i < dim; i += 128) {
    float v = (x[i] - mu) * inv * g[i] + b[i];
    size_t off = (size_t)row * dim + i;
    if (outf)  outf[off]  = v;
    if (outbf) outbf[off] = f2bf(v);
  }
}

// Single-query cross-attention per batch: scores over 196 keys, softmax, o = a@v
__global__ __launch_bounds__(256) void attn_readout(
    const float* __restrict__ q, const float* __restrict__ kv, u16* __restrict__ obf)
{
  const int b = blockIdx.x;
  const int tid = threadIdx.x;
  __shared__ float sc[TOK];
  __shared__ float red[256];
  __shared__ float s_max, s_sum;
  const float* qr = q + (size_t)b * DIM;

  for (int t = tid; t < TOK; t += 256) {
    const float* kr = kv + (size_t)(b * TOK + t) * (2 * DIM);
    float s = 0.f;
    for (int d = 0; d < DIM; ++d) s += qr[d] * kr[d];
    sc[t] = s * 0.05103103630798288f;      // 384^-0.5
  }
  __syncthreads();

  float m = -3.4e38f;
  for (int t = tid; t < TOK; t += 256) m = fmaxf(m, sc[t]);
  red[tid] = m; __syncthreads();
  for (int o = 128; o > 0; o >>= 1) { if (tid < o) red[tid] = fmaxf(red[tid], red[tid + o]); __syncthreads(); }
  if (tid == 0) s_max = red[0];
  __syncthreads();
  const float mx = s_max;

  float s = 0.f;
  for (int t = tid; t < TOK; t += 256) { float e = __expf(sc[t] - mx); sc[t] = e; s += e; }
  red[tid] = s; __syncthreads();
  for (int o = 128; o > 0; o >>= 1) { if (tid < o) red[tid] += red[tid + o]; __syncthreads(); }
  if (tid == 0) s_sum = red[0];
  __syncthreads();
  const float inv = 1.0f / s_sum;

  for (int d = tid; d < DIM; d += 256) {
    float o = 0.f;
    for (int t = 0; t < TOK; ++t)
      o += sc[t] * kv[(size_t)(b * TOK + t) * (2 * DIM) + DIM + d];
    obf[(size_t)b * DIM + d] = f2bf(o * inv);
  }
}

// logits (32x1000) = lat(32x384) @ head_w(384x1000) + head_b   (tiny; fp32 VALU)
__global__ __launch_bounds__(256) void head_proj(
    const float* __restrict__ lat, const float* __restrict__ w,
    const float* __restrict__ bias, float* __restrict__ out)
{
  int idx = blockIdx.x * 256 + threadIdx.x;
  if (idx >= BATCH * NCLS) return;
  int b = idx / NCLS, n = idx % NCLS;
  float s = bias[n];
  const float* l = lat + (size_t)b * DIM;
  for (int k = 0; k < DIM; ++k) s += l[k] * w[(size_t)k * NCLS + n];
  out[idx] = s;
}

// ---------------------------------------------------------------------------
// Host orchestration
// ---------------------------------------------------------------------------
extern "C" void kernel_launch(void* const* d_in, const int* in_sizes, int n_in,
                              void* d_out, int out_size, void* d_ws, size_t ws_size,
                              hipStream_t stream) {
  (void)in_sizes; (void)n_in; (void)out_size; (void)ws_size;
  const float* x        = (const float*)d_in[0];
  const float* stem_w   = (const float*)d_in[1];
  const float* stem_b   = (const float*)d_in[2];
  const float* blk_dw_w = (const float*)d_in[3];
  const float* blk_dw_b = (const float*)d_in[4];
  const float* blk_ln_g = (const float*)d_in[5];
  const float* blk_ln_b = (const float*)d_in[6];
  const float* blk_w1   = (const float*)d_in[7];
  const float* blk_b1   = (const float*)d_in[8];
  const float* blk_w2   = (const float*)d_in[9];
  const float* blk_b2   = (const float*)d_in[10];
  const float* q_w      = (const float*)d_in[11];
  const float* kv_w     = (const float*)d_in[12];
  const float* out_w    = (const float*)d_in[13];
  const float* out_b    = (const float*)d_in[14];
  const float* latent   = (const float*)d_in[15];
  const float* lln_g    = (const float*)d_in[16];
  const float* lln_b    = (const float*)d_in[17];
  const float* head_w   = (const float*)d_in[18];
  const float* head_b   = (const float*)d_in[19];
  float* logits = (float*)d_out;

  // workspace carve-up (256B aligned)
  unsigned char* ws = (unsigned char*)d_ws;
  size_t off = 0;
  auto alloc = [&](size_t bytes) -> void* {
    void* p = ws + off;
    off = (off + bytes + 255) & ~(size_t)255;
    return p;
  };
  u16*   patches = (u16*)  alloc((size_t)NTOK * PATCH_K * 2);
  u16*   stemwbf = (u16*)  alloc((size_t)DIM * PATCH_K * 2);
  float* hbuf    = (float*)alloc((size_t)NTOK * DIM * 4);
  u16*   ctxbf   = (u16*)  alloc((size_t)NTOK * DIM * 2);
  float* ybuf    = (float*)alloc((size_t)NTOK * DIM * 4);
  u16*   abuf    = (u16*)  alloc((size_t)NTOK * DIM * 2);
  u16*   midbf   = (u16*)  alloc((size_t)NTOK * HIDDEN * 2);
  u16*   w1t     = (u16*)  alloc((size_t)HIDDEN * DIM * 2);
  u16*   w2t     = (u16*)  alloc((size_t)DIM * HIDDEN * 2);
  u16*   kvwt    = (u16*)  alloc((size_t)(2 * DIM) * DIM * 2);
  u16*   qwt     = (u16*)  alloc((size_t)DIM * DIM * 2);
  u16*   outwt   = (u16*)  alloc((size_t)DIM * DIM * 2);
  float* kvbuf   = (float*)alloc((size_t)NTOK * (2 * DIM) * 4);
  float* latf    = (float*)alloc((size_t)LATPAD * DIM * 4);
  u16*   latbf   = (u16*)  alloc((size_t)LATPAD * DIM * 2);
  float* lattmp  = (float*)alloc((size_t)LATPAD * DIM * 4);
  float* qbuf    = (float*)alloc((size_t)LATPAD * DIM * 4);
  u16*   obf     = (u16*)  alloc((size_t)LATPAD * DIM * 2);

  auto blks = [](int n) { return (n + 255) / 256; };
  const int MT = NTOK / 128;   // 49 M-blocks (128 rows each)

  // ---- weight conversions (re-done each call; cheap, deterministic) ----
  convert_f32_bf16<<<blks(DIM * PATCH_K), 256, 0, stream>>>(stem_w, stemwbf, DIM * PATCH_K); // already N-major
  transpose_f32_bf16<<<blks(DIM * DIM),     256, 0, stream>>>(q_w,   qwt,   DIM, DIM);
  transpose_f32_bf16<<<blks(DIM * 2 * DIM), 256, 0, stream>>>(kv_w,  kvwt,  DIM, 2 * DIM);
  transpose_f32_bf16<<<blks(DIM * DIM),     256, 0, stream>>>(out_w, outwt, DIM, DIM);
  patchify<<<blks(NTOK * PATCH_K), 256, 0, stream>>>(x, patches);
  init_latent<<<blks(LATPAD * DIM), 256, 0, stream>>>(latent, latf, latbf);
  hipMemsetAsync(obf, 0, (size_t)LATPAD * DIM * 2, stream);   // pad rows for out-proj GEMM

  // ---- stem: h = patches @ stem_w^T + stem_b  (writes fp32 h + bf16 ctx) ----
  gemm_bf16_wmma<<<dim3(MT, DIM / 64), 256, 0, stream>>>(
      patches, stemwbf, hbuf, ctxbf, stem_b, nullptr, NTOK, DIM, PATCH_K, /*bias*/1);

  // ---- readout: lat = LN(lat + digup(lat, ctx)) ----
  auto readout = [&]() {
    // kv = ctx @ kv_w   (6272 x 768)
    gemm_bf16_wmma<<<dim3(MT, (2 * DIM) / 64), 256, 0, stream>>>(
        ctxbf, kvwt, kvbuf, nullptr, nullptr, nullptr, NTOK, 2 * DIM, DIM, 0);
    // q = lat @ q_w    (padded to 64 rows; waves beyond M clamp + skip epilogue)
    gemm_bf16_wmma<<<dim3(1, DIM / 64), 256, 0, stream>>>(
        latbf, qwt, qbuf, nullptr, nullptr, nullptr, LATPAD, DIM, DIM, 0);
    // softmax attention over 196 keys per batch
    attn_readout<<<BATCH, 256, 0, stream>>>(qbuf, kvbuf, obf);
    // lat_tmp = lat + o @ out_w + out_b
    gemm_bf16_wmma<<<dim3(1, DIM / 64), 256, 0, stream>>>(
        obf, outwt, lattmp, nullptr, out_b, latf, LATPAD, DIM, DIM, /*bias|res*/1 | 4);
    // lat = LN(lat_tmp), emit fp32 + bf16 (bf16 feeds next q GEMM)
    ln_rows<<<BATCH, 128, 0, stream>>>(lattmp, lln_g, lln_b, 1e-5f, latf, latbf, DIM);
  };
  readout();

  // ---- 18 blocks ----
  for (int d = 0; d < DEPTH; ++d) {
    const float* dw_w = blk_dw_w + (size_t)d * DIM * 49;
    const float* dw_b = blk_dw_b + (size_t)d * DIM;
    const float* lg   = blk_ln_g + (size_t)d * DIM;
    const float* lb   = blk_ln_b + (size_t)d * DIM;
    const float* w1   = blk_w1   + (size_t)d * DIM * HIDDEN;
    const float* b1   = blk_b1   + (size_t)d * HIDDEN;
    const float* w2   = blk_w2   + (size_t)d * HIDDEN * DIM;
    const float* b2   = blk_b2   + (size_t)d * DIM;

    dwconv7<<<blks(NTOK * DIM), 256, 0, stream>>>(hbuf, dw_w, dw_b, ybuf);
    ln_rows<<<NTOK, 128, 0, stream>>>(ybuf, lg, lb, 1e-6f, nullptr, abuf, DIM);

    transpose_f32_bf16<<<blks(DIM * HIDDEN), 256, 0, stream>>>(w1, w1t, DIM, HIDDEN);
    // mid = gelu(ln @ w1 + b1) -> bf16 only
    gemm_bf16_wmma<<<dim3(MT, HIDDEN / 64), 256, 0, stream>>>(
        abuf, w1t, nullptr, midbf, b1, nullptr, NTOK, HIDDEN, DIM, /*bias|gelu*/1 | 2);

    transpose_f32_bf16<<<blks(HIDDEN * DIM), 256, 0, stream>>>(w2, w2t, HIDDEN, DIM);
    // h = h + mid @ w2 + b2 -> fp32 h + bf16 ctx, fused residual
    gemm_bf16_wmma<<<dim3(MT, DIM / 64), 256, 0, stream>>>(
        midbf, w2t, hbuf, ctxbf, b2, hbuf, NTOK, DIM, HIDDEN, /*bias|res*/1 | 4);

    readout();
  }

  // ---- head ----
  head_proj<<<blks(BATCH * NCLS), 256, 0, stream>>>(latf, head_w, head_b, logits);
}
